// LSTM_20590073217386
// MI455X (gfx1250) — matrix operand order repaired and checked by reading, not compiled, not run on hardware
//
#include <hip/hip_runtime.h>
#include <math.h>

typedef __attribute__((ext_vector_type(16))) __bf16 v16bf;
typedef __attribute__((ext_vector_type(8)))  __bf16 v8bf;
typedef __attribute__((ext_vector_type(8)))  float  v8f;

#define BATCH 2048
#define TSTEPS 336
#define FIN 10

// Branch-free fast activations: single v_exp_f32 + single v_rcp_f32 each,
// no EXEC divergence and no IEEE-division fixup chains in the hot loop.
__device__ __forceinline__ float fast_rcp(float x) {
    return __builtin_amdgcn_rcpf(x);
}
__device__ __forceinline__ float sigm(float x) {
    return fast_rcp(1.0f + __expf(-x));
}
__device__ __forceinline__ float tanh_fast(float x) {
    float e = __expf(-2.0f * fabsf(x));
    float t = (1.0f - e) * fast_rcp(1.0f + e);
    return copysignf(t, x);
}

// ---------------------------------------------------------------------------
// Recurrent bidirectional LSTM layer. One wave owns 16 batch rows and runs the
// full time loop. Gates layout (N): [i(HP) f(HP) g(HP) o(HP)], K layout:
// [h(HP) | x(XN) | 1.0(bias col) | pad]. All matrices bf16 in LDS, f32 WMMA
// accumulation; bias folded into weight column K = HP+XN so the WMMA C input
// is an inline zero (no per-step bias splat).
// Gate WMMAs run kt-major over 4 independent accumulators so consecutive
// WMMAs never depend on each other (hides LDS weight-load latency).
//   H  = true hidden, HP = hidden padded to mult of 16
//   XN = input features, KP = K padded to mult of 32 (KP >= HP+XN+1)
//   NTILES = 4*HP/16, KTILES = KP/32
//   LAYER==1 reads f32 x [B,T,FIN]; LAYER==2 reads bf16 h1 [T,B,140]
// ---------------------------------------------------------------------------
template<int H, int HP, int XN, int KP, int NTILES, int KTILES, int LAYER>
__global__ void lstm_bidir_kernel(
    const float*  __restrict__ xf,
    const __bf16* __restrict__ xb,
    const float* __restrict__ wih_f, const float* __restrict__ whh_f,
    const float* __restrict__ bih_f, const float* __restrict__ bhh_f,
    const float* __restrict__ wih_b, const float* __restrict__ whh_b,
    const float* __restrict__ bih_b, const float* __restrict__ bhh_b,
    __bf16* __restrict__ hout)   // [T, B, 2*H] bf16, [fwd | bwd] concat
{
    constexpr int HT = HP / 16;                    // hidden tiles
    constexpr int WELEMS = NTILES * KTILES * 512;  // 32 lanes * 16 elems per tile-pair

    const int tid  = threadIdx.x;
    const int lane = tid & 31;
    const int wv   = tid >> 5;                  // wave in block (2 waves)
    const int dir  = blockIdx.y;                // 0=fwd, 1=bwd
    const int b0   = blockIdx.x * 32 + wv * 16; // batch row base of this wave
    const int ln   = lane & 15;
    const int hfl  = lane >> 4;                 // lane half

    const float* wih = dir ? wih_b : wih_f;
    const float* whh = dir ? whh_b : whh_f;
    const float* bih = dir ? bih_b : bih_f;
    const float* bhh = dir ? bhh_b : bhh_f;

    extern __shared__ char smem[];
    __bf16* ldsW = (__bf16*)smem;                                  // WELEMS bf16
    __bf16* ldsA = (__bf16*)(smem + (size_t)WELEMS * 2) + wv * 16 * KP;

    // ---- repack weights into exact B-operand lane layout (bf16) ----
    // operand for (nt,kt): 32 lanes * 16 contiguous bf16; lane<16 -> N=nt*16+lane,
    // K=kt*32 + e ; lane>=16 -> N=nt*16+lane-16, K=kt*32+16+e  (e packed 2/VGPR)
    // K < HP           : W_hh column K        (valid K < H)
    // HP <= K < HP+XN  : W_ih column K-HP
    // K == HP+XN       : bias column (b_ih + b_hh), A supplies constant 1.0
    for (int i = tid; i < WELEMS; i += blockDim.x) {
        int e  = i & 15;
        int l  = (i >> 4) & 31;
        int kt = (i >> 9) % KTILES;
        int nt = (i >> 9) / KTILES;
        int N = nt * 16 + (l & 15);
        int K = kt * 32 + (l >> 4) * 16 + e;
        int gate = N / HP, unit = N % HP;
        float v = 0.0f;
        if (unit < H) {
            int row = gate * H + unit;
            if (K < HP) { if (K < H)  v = whh[row * H + K]; }
            else {
                int c2 = K - HP;
                if      (c2 < XN)  v = wih[row * XN + c2];
                else if (c2 == XN) v = bih[row] + bhh[row];
            }
        }
        ldsW[i] = (__bf16)v;
    }
    __syncthreads();

    // ---- zero this wave's A staging buffer (h part + pad) ----
    for (int i = lane; i < 16 * KP; i += 32) ldsA[i] = (__bf16)0.0f;

    v8f c[HT];
    #pragma unroll
    for (int h = 0; h < HT; ++h)
        #pragma unroll
        for (int e = 0; e < 8; ++e) c[h][e] = 0.0f;

    v8f czero;                      // loop-invariant WMMA C input (folds to inline 0)
    #pragma unroll
    for (int e = 0; e < 8; ++e) czero[e] = 0.0f;

    // weight operand load helper (nt, kt) -> v16bf
    const __bf16* ldsWlane = ldsW + (size_t)lane * 16;
    auto loadW = [&](int nt, int kt) -> v16bf {
        return *(const v16bf*)(ldsWlane + (size_t)(nt * KTILES + kt) * 512);
    };

    for (int ti = 0; ti < TSTEPS; ++ti) {
        const int t = dir ? (TSTEPS - 1 - ti) : ti;

        // ---- stage x (+ bias 1.0 column) into A buffer, K range [HP, KP) ----
        if constexpr (LAYER == 1) {
            // 16 pad slots per row: x in [0,XN), 1.0 at XN, zeros after
            for (int i = lane; i < 16 * 16; i += 32) {
                int r = i >> 4, k = i & 15;
                float v = (k < XN) ? xf[(size_t)(b0 + r) * TSTEPS * FIN + t * FIN + k]
                                   : (k == XN ? 1.0f : 0.0f);
                ldsA[r * KP + HP + k] = (__bf16)v;
            }
        } else {
            constexpr int XWU = (KP - HP) / 2;   // uints per row in x region
            constexpr int XVU = XN / 2;          // valid uints (XN even)
            const unsigned* src = (const unsigned*)(xb + (size_t)t * BATCH * XN);
            for (int i = lane; i < 16 * XWU; i += 32) {
                int r = i / XWU, ku = i % XWU;
                unsigned v = (ku < XVU) ? src[(size_t)(b0 + r) * XVU + ku]
                                        : (ku == XVU ? 0x3F80u : 0u); // {1.0bf16, 0}
                ((unsigned*)(ldsA + r * KP + HP))[ku] = v;
            }
        }
        asm volatile("" ::: "memory");

        // ---- load A operands (16-bit A layout: two 16B chunks per lane) ----
        v16bf a[KTILES];
        #pragma unroll
        for (int kt = 0; kt < KTILES; ++kt) {
            const __bf16* p = ldsA + ln * KP + kt * 32 + hfl * 16;
            v8bf lo = *(const v8bf*)p;
            v8bf hi = *(const v8bf*)(p + 8);
            a[kt] = __builtin_shufflevector(lo, hi, 0,1,2,3,4,5,6,7,8,9,10,11,12,13,14,15);
        }

        // ---- per hidden tile: 4 gate tiles (i,f,g,o) -> activations -> h ----
        #pragma unroll
        for (int h = 0; h < HT; ++h) {
            // kt-major over 4 independent accumulators: consecutive WMMAs are
            // independent, so weight-load waits overlap WMMA execution.
            v8f gq[4];
            #pragma unroll
            for (int q = 0; q < 4; ++q) gq[q] = czero;
            #pragma unroll
            for (int kt = 0; kt < KTILES; ++kt) {
                #pragma unroll
                for (int q = 0; q < 4; ++q) {
                    v16bf w = loadW(q * HT + h, kt);
                    gq[q] = __builtin_amdgcn_wmma_f32_16x16x32_bf16(
                                false, a[kt], false, w, (short)0, gq[q], false, false);
                }
            }
            const int unit = h * 16 + ln;
            #pragma unroll
            for (int e = 0; e < 8; ++e) {
                float iv = sigm(gq[0][e]);
                float fv = sigm(gq[1][e]);
                float gv = tanh_fast(gq[2][e]);
                float ov = sigm(gq[3][e]);
                float cv = fv * c[h][e] + iv * gv;
                c[h][e] = cv;
                float hv = ov * tanh_fast(cv);
                int m = e + hfl * 8;                     // D layout: row m, col = unit
                ldsA[m * KP + unit] = (__bf16)hv;        // feeds next step's A (h part)
                if (unit < H)
                    hout[((size_t)t * BATCH + b0 + m) * (2 * H) + dir * H + unit] = (__bf16)hv;
            }
        }
        asm volatile("" ::: "memory");
    }
}

// ---------------------------------------------------------------------------
// Dense head: [T,B,42] bf16 -> relu(42x30) -> relu(30x20) -> 20x1 -> out[B,T]
// ---------------------------------------------------------------------------
__global__ void dense_head_kernel(
    const __bf16* __restrict__ h2,
    const float* __restrict__ wd1, const float* __restrict__ bd1,
    const float* __restrict__ wd2, const float* __restrict__ bd2,
    const float* __restrict__ wo,  const float* __restrict__ bo,
    float* __restrict__ out)
{
    __shared__ float s1[30 * 42], sb1[30], s2[20 * 30], sb2[20], so[20], sbo[1];
    const int tid = threadIdx.x;
    for (int i = tid; i < 30 * 42; i += blockDim.x) s1[i] = wd1[i];
    for (int i = tid; i < 30;      i += blockDim.x) sb1[i] = bd1[i];
    for (int i = tid; i < 20 * 30; i += blockDim.x) s2[i] = wd2[i];
    for (int i = tid; i < 20;      i += blockDim.x) sb2[i] = bd2[i];
    for (int i = tid; i < 20;      i += blockDim.x) so[i] = wo[i];
    if (tid == 0) sbo[0] = bo[0];
    __syncthreads();

    const int gid = blockIdx.x * blockDim.x + tid;
    if (gid >= BATCH * TSTEPS) return;
    const int b = gid / TSTEPS, t = gid % TSTEPS;
    const __bf16* hp = h2 + ((size_t)t * BATCH + b) * 42;

    float v[42];
    #pragma unroll
    for (int k = 0; k < 42; ++k) v[k] = (float)hp[k];

    float r1[30];
    #pragma unroll
    for (int j = 0; j < 30; ++j) {
        float s = sb1[j];
        #pragma unroll
        for (int k = 0; k < 42; ++k) s += s1[j * 42 + k] * v[k];
        r1[j] = fmaxf(s, 0.0f);
    }
    float r2[20];
    #pragma unroll
    for (int j = 0; j < 20; ++j) {
        float s = sb2[j];
        #pragma unroll
        for (int k = 0; k < 30; ++k) s += s2[j * 30 + k] * r1[k];
        r2[j] = fmaxf(s, 0.0f);
    }
    float y = sbo[0];
    #pragma unroll
    for (int k = 0; k < 20; ++k) y += so[k] * r2[k];
    out[gid] = y;   // gid == b*T + t  => [B,T,1]
}

extern "C" void kernel_launch(void* const* d_in, const int* in_sizes, int n_in,
                              void* d_out, int out_size, void* d_ws, size_t ws_size,
                              hipStream_t stream)
{
    (void)in_sizes; (void)n_in; (void)out_size; (void)ws_size;
    const float* x      = (const float*)d_in[0];
    const float* w1f_ih = (const float*)d_in[1];
    const float* w1f_hh = (const float*)d_in[2];
    const float* b1f_ih = (const float*)d_in[3];
    const float* b1f_hh = (const float*)d_in[4];
    const float* w1b_ih = (const float*)d_in[5];
    const float* w1b_hh = (const float*)d_in[6];
    const float* b1b_ih = (const float*)d_in[7];
    const float* b1b_hh = (const float*)d_in[8];
    const float* w2f_ih = (const float*)d_in[9];
    const float* w2f_hh = (const float*)d_in[10];
    const float* b2f_ih = (const float*)d_in[11];
    const float* b2f_hh = (const float*)d_in[12];
    const float* w2b_ih = (const float*)d_in[13];
    const float* w2b_hh = (const float*)d_in[14];
    const float* b2b_ih = (const float*)d_in[15];
    const float* b2b_hh = (const float*)d_in[16];
    const float* wd1    = (const float*)d_in[17];
    const float* bd1    = (const float*)d_in[18];
    const float* wd2    = (const float*)d_in[19];
    const float* bd2    = (const float*)d_in[20];
    const float* wo     = (const float*)d_in[21];
    const float* bo     = (const float*)d_in[22];

    // workspace: h1 [T,B,140] bf16, then h2 [T,B,42] bf16
    __bf16* h1 = (__bf16*)d_ws;
    __bf16* h2 = (__bf16*)((char*)d_ws + (size_t)TSTEPS * BATCH * 140 * 2);

    // layer 1: H=70 HP=80 XN=10 KP=96 NTILES=20 KTILES=3  (bias col K=90)
    {
        constexpr size_t lds = (size_t)20 * 3 * 512 * 2 + 2 * 16 * 96 * 2; // 67584
        lstm_bidir_kernel<70, 80, 10, 96, 20, 3, 1>
            <<<dim3(BATCH / 32, 2), 64, lds, stream>>>(
                x, (const __bf16*)nullptr,
                w1f_ih, w1f_hh, b1f_ih, b1f_hh,
                w1b_ih, w1b_hh, b1b_ih, b1b_hh, h1);
    }
    // layer 2: H=21 HP=32 XN=140 KP=192 NTILES=8 KTILES=6 (bias col K=172)
    {
        constexpr size_t lds = (size_t)8 * 6 * 512 * 2 + 2 * 16 * 192 * 2; // 61440
        lstm_bidir_kernel<21, 32, 140, 192, 8, 6, 2>
            <<<dim3(BATCH / 32, 2), 64, lds, stream>>>(
                (const float*)nullptr, h1,
                w2f_ih, w2f_hh, b2f_ih, b2f_hh,
                w2b_ih, w2b_hh, b2b_ih, b2b_hh, h2);
    }
    // dense head
    dense_head_kernel<<<(BATCH * TSTEPS + 255) / 256, 256, 0, stream>>>(
        h2, wd1, bd1, wd2, bd2, wo, bo, (float*)d_out);
}
